// KnowledgeGroundedDecoder_75204877353155
// MI455X (gfx1250) — compile-verified, matrix operand order checked by reading, not run on hardware
//
#include <hip/hip_runtime.h>
#include <hip/hip_bf16.h>

// ---------------------------------------------------------------------------
// KnowledgeGroundedDecoder for MI455X (gfx1250, wave32, WMMA).
// B=8 L=128 V=50257 E=768 M=400 Mt=800 R=40, NUM_HOPS=2, GAMMA=0.8
// ---------------------------------------------------------------------------

#define B_   8
#define L_   128
#define V_   50257
#define E_   768
#define M_   400
#define MT_  800
#define BL_  (B_ * L_)

typedef __attribute__((ext_vector_type(16))) _Float16 v16h;
typedef __attribute__((ext_vector_type(8)))  _Float16 v8h;
typedef __attribute__((ext_vector_type(8)))  float    v8f;

#define WMMA_F16(a, b, c) \
  __builtin_amdgcn_wmma_f32_16x16x32_f16(false, (a), false, (b), (short)0, (c), false, false)

// 16x32 fragment from a row-major f16 row: the ISA layout makes each lane's
// 16 halves two contiguous 8-half (16B) runs -> two b128 loads, no VALU cvt.
__device__ __forceinline__ v16h frag_f16(const _Float16* __restrict__ rowp, int k0, int sel) {
  const v8h lo = *(const v8h*)(rowp + k0 + sel * 8);
  const v8h hi = *(const v8h*)(rowp + k0 + 16 + sel * 8);
  return __builtin_shufflevector(lo, hi, 0, 1, 2, 3, 4, 5, 6, 7,
                                 8, 9, 10, 11, 12, 13, 14, 15);
}

// Converting loader for operands that must stay f32 (atomic-built buffers).
__device__ __forceinline__ v16h frag_f32(const float* __restrict__ rowp, int k0, int sel) {
  v16h f;
#pragma unroll
  for (int j = 0; j < 8; ++j) {
    const int kp = (j >> 2) * 16 + sel * 8 + (j & 3) * 2;
    const float2 d = *(const float2*)(rowp + k0 + kp);
    f[2 * j]     = (_Float16)d.x;
    f[2 * j + 1] = (_Float16)d.y;
  }
  return f;
}

// ---------------------------------------------------------------------------
// Utility kernels
// ---------------------------------------------------------------------------
__global__ void k_zero(float* __restrict__ p, int n) {
  int i = blockIdx.x * 256 + threadIdx.x;
  if (i < n) p[i] = 0.0f;
}

__global__ void k_cvt(const float* __restrict__ src, _Float16* __restrict__ dst, int n) {
  int i = blockIdx.x * 256 + threadIdx.x;
  if (i < n) dst[i] = (_Float16)src[i];
}

// dst[r, :] = src[ids[r], :]   (E=768, float4 granularity: 192 per row)
__global__ void k_gather(const float* __restrict__ src, const int* __restrict__ ids,
                         float* __restrict__ dst, int nrows) {
  int gid = blockIdx.x * 256 + threadIdx.x;
  if (gid >= nrows * 192) return;
  int r = gid / 192, e = gid % 192;
  const float4* s = (const float4*)(src + (size_t)ids[r] * E_);
  ((float4*)(dst + (size_t)r * E_))[e] = s[e];
}

// GCN degree counts (head+tail) and multi-hop tail counts
__global__ void k_cnt(const int* __restrict__ head, const int* __restrict__ tail,
                      float* __restrict__ cntg, float* __restrict__ cntm) {
  int gid = blockIdx.x * 256 + threadIdx.x;
  if (gid >= B_ * MT_) return;
  int b = gid / MT_;
  atomicAdd(&cntg[b * M_ + head[gid]], 1.0f);
  atomicAdd(&cntg[b * M_ + tail[gid]], 1.0f);
  atomicAdd(&cntm[b * M_ + tail[gid]], 1.0f);
}

// upd[tail] += cr[head]-rr ; upd[head] += cr[tail]-rr   (per element)
__global__ void k_scatter(const float* __restrict__ cr, const float* __restrict__ rr,
                          const int* __restrict__ head, const int* __restrict__ tail,
                          float* __restrict__ upd) {
  int gid = blockIdx.x * 256 + threadIdx.x;
  if (gid >= B_ * MT_ * E_) return;
  int e  = gid % E_;
  int bt = gid / E_;       // b*Mt + t
  int b  = bt / MT_;
  int h = head[bt], t = tail[bt];
  float rv = rr[(size_t)bt * E_ + e];
  float ch = cr[((size_t)b * M_ + h) * E_ + e];
  float ct = cr[((size_t)b * M_ + t) * E_ + e];
  atomicAdd(&upd[((size_t)b * M_ + t) * E_ + e], ch - rv);
  atomicAdd(&upd[((size_t)b * M_ + h) * E_ + e], ct - rv);
}

// ---------------------------------------------------------------------------
// WMMA GEMM kernels (one wave == one 16x16 C tile, K stepped by 32)
// ---------------------------------------------------------------------------

// node_h = f16(relu(cr @ Ws1^T + (upd @ Wn1^T)/clip(cnt,1)))
__global__ void k_node(const float* __restrict__ cr, const float* __restrict__ upd,
                       const _Float16* __restrict__ Ws1, const _Float16* __restrict__ Wn1,
                       const float* __restrict__ cnt, _Float16* __restrict__ node) {
  const int lane = threadIdx.x, rc = lane & 15, sel = lane >> 4;
  const int m0 = blockIdx.x * 16, o0 = blockIdx.y * 16, b = blockIdx.z;
  const float* a1 = cr  + ((size_t)b * M_ + m0 + rc) * E_;
  const float* a2 = upd + ((size_t)b * M_ + m0 + rc) * E_;
  const _Float16* b1 = Ws1 + (size_t)(o0 + rc) * E_;
  const _Float16* b2 = Wn1 + (size_t)(o0 + rc) * E_;
  v8f s = {}, n = {};
  for (int k0 = 0; k0 < E_; k0 += 32) {
    s = WMMA_F16(frag_f32(a1, k0, sel), frag_f16(b1, k0, sel), s);
    n = WMMA_F16(frag_f32(a2, k0, sel), frag_f16(b2, k0, sel), n);
  }
#pragma unroll
  for (int r = 0; r < 8; ++r) {
    int m = m0 + r + 8 * sel;
    float c = cnt[b * M_ + m];
    c = c < 1.0f ? 1.0f : c;
    float v = s[r] + n[r] / c;
    node[((size_t)b * M_ + m) * E_ + o0 + rc] = (_Float16)(v > 0.0f ? v : 0.0f);
  }
}

// relh_h = f16(rr @ Wr1^T)
__global__ void k_relh(const float* __restrict__ rr, const _Float16* __restrict__ Wr1,
                       _Float16* __restrict__ relh) {
  const int lane = threadIdx.x, rc = lane & 15, sel = lane >> 4;
  const int t0 = blockIdx.x * 16, o0 = blockIdx.y * 16, b = blockIdx.z;
  const float* ap = rr + ((size_t)b * MT_ + t0 + rc) * E_;
  const _Float16* bp = Wr1 + (size_t)(o0 + rc) * E_;
  v8f acc = {};
  for (int k0 = 0; k0 < E_; k0 += 32)
    acc = WMMA_F16(frag_f32(ap, k0, sel), frag_f16(bp, k0, sel), acc);
#pragma unroll
  for (int r = 0; r < 8; ++r)
    relh[((size_t)b * MT_ + t0 + r + 8 * sel) * E_ + o0 + rc] = (_Float16)acc[r];
}

// tl_h = f16(node[head]@Wt[:,0:768]^T + relh@Wt[:,768:1536]^T + node[tail]@Wt[:,1536:2304]^T)
__global__ void k_tl(const _Float16* __restrict__ node, const _Float16* __restrict__ relh,
                     const int* __restrict__ head, const int* __restrict__ tail,
                     const _Float16* __restrict__ Wt, _Float16* __restrict__ tl) {
  const int lane = threadIdx.x, rc = lane & 15, sel = lane >> 4;
  const int t0 = blockIdx.x * 16, o0 = blockIdx.y * 16, b = blockIdx.z;
  const int t = t0 + rc;
  const int h  = head[b * MT_ + t];
  const int ta = tail[b * MT_ + t];
  const _Float16* ah = node + ((size_t)b * M_  + h)  * E_;
  const _Float16* ar = relh + ((size_t)b * MT_ + t)  * E_;
  const _Float16* at = node + ((size_t)b * M_  + ta) * E_;
  const _Float16* bh = Wt + (size_t)(o0 + rc) * (3 * E_);
  const _Float16* br = bh + E_;
  const _Float16* bt = bh + 2 * E_;
  v8f acc = {};
  for (int k0 = 0; k0 < E_; k0 += 32) {
    acc = WMMA_F16(frag_f16(ah, k0, sel), frag_f16(bh, k0, sel), acc);
    acc = WMMA_F16(frag_f16(ar, k0, sel), frag_f16(br, k0, sel), acc);
    acc = WMMA_F16(frag_f16(at, k0, sel), frag_f16(bt, k0, sel), acc);
  }
#pragma unroll
  for (int r = 0; r < 8; ++r)
    tl[((size_t)b * MT_ + t0 + r + 8 * sel) * E_ + o0 + rc] = (_Float16)acc[r];
}

// triple_prob[b,l,m] = sigmoid(hidden[b,l,:] . tl[b,m,:]) * (labels != -1)
__global__ void k_tp(const _Float16* __restrict__ hidden, const _Float16* __restrict__ tl,
                     const int* __restrict__ lab, float* __restrict__ tp) {
  const int lane = threadIdx.x, rc = lane & 15, sel = lane >> 4;
  const int m0 = blockIdx.x * 16, l0 = blockIdx.y * 16, b = blockIdx.z;
  const _Float16* ap = hidden + ((size_t)b * L_  + l0 + rc) * E_;
  const _Float16* bp = tl     + ((size_t)b * MT_ + m0 + rc) * E_;
  v8f acc = {};
  for (int k0 = 0; k0 < E_; k0 += 32)
    acc = WMMA_F16(frag_f16(ap, k0, sel), frag_f16(bp, k0, sel), acc);
  const int m = m0 + rc;
  const float mask = (lab[b * MT_ + m] != -1) ? 1.0f : 0.0f;
#pragma unroll
  for (int r = 0; r < 8; ++r) {
    float sg = 1.0f / (1.0f + __expf(-acc[r]));
    tp[((size_t)b * L_ + l0 + r + 8 * sel) * MT_ + m] = sg * mask;
  }
}

// logits[row, v] = hidden[row,:] . wte[v,:]  -> staged into d_out.
// 4 waves / block: 4 adjacent vocab tiles share the A-fragment stream (wte_h
// is 77MB f16, L2-resident across the 64 row-tile passes).
__global__ void k_logits(const _Float16* __restrict__ hidden, const _Float16* __restrict__ wte,
                         float* __restrict__ out) {
  const int lane = threadIdx.x & 31, w = threadIdx.x >> 5;
  const int rc = lane & 15, sel = lane >> 4;
  const int v0 = (blockIdx.x * 4 + w) * 16;
  const int r0 = blockIdx.y * 16;
  const _Float16* ap = hidden + (size_t)(r0 + rc) * E_;
  int vrow = v0 + rc;
  if (vrow > V_ - 1) vrow = V_ - 1;                 // clamp loads at the ragged edge
  const _Float16* bp = wte + (size_t)vrow * E_;
  v8f acc = {};
  for (int k0 = 0; k0 < E_; k0 += 32) {
    __builtin_prefetch(bp + k0 + 256, 0, 1);        // global_prefetch_b8
    acc = WMMA_F16(frag_f16(ap, k0, sel), frag_f16(bp, k0, sel), acc);
  }
  if (v0 + rc < V_) {
#pragma unroll
    for (int r = 0; r < 8; ++r)
      out[(size_t)(r0 + r + 8 * sel) * V_ + v0 + rc] = acc[r];
  }
}

// ---------------------------------------------------------------------------
// Softmax stats over V per (b,l) row
// ---------------------------------------------------------------------------
__global__ void k_stats(const float* __restrict__ logits, float* __restrict__ stats) {
  __shared__ float red[256];
  const int row = blockIdx.x, tid = threadIdx.x;
  const float* p = logits + (size_t)row * V_;
  float mx = -1e30f;
  for (int v = tid; v < V_; v += 256) mx = fmaxf(mx, p[v]);
  red[tid] = mx; __syncthreads();
  for (int s = 128; s > 0; s >>= 1) { if (tid < s) red[tid] = fmaxf(red[tid], red[tid + s]); __syncthreads(); }
  mx = red[0]; __syncthreads();
  float sum = 0.0f;
  for (int v = tid; v < V_; v += 256) sum += expf(p[v] - mx);
  red[tid] = sum; __syncthreads();
  for (int s = 128; s > 0; s >>= 1) { if (tid < s) red[tid] += red[tid + s]; __syncthreads(); }
  if (tid == 0) { stats[row * 2] = mx; stats[row * 2 + 1] = red[0]; }
}

__global__ void k_gate(const float* __restrict__ hidden, const float* __restrict__ gw,
                       const float* __restrict__ gb, float* __restrict__ gate) {
  int row = blockIdx.x * blockDim.x + threadIdx.x;
  if (row >= BL_) return;
  const float* h = hidden + (size_t)row * E_;
  float s = gb[0];
  for (int e = 0; e < E_; ++e) s += h[e] * gw[e];
  gate[row] = 1.0f / (1.0f + expf(-s));
}

// ---------------------------------------------------------------------------
// 2-hop scatter-mean propagation + softmax over M, one block per (b,l) row.
// Uses LDS atomics (ds_add) for the per-tail accumulation.
// ---------------------------------------------------------------------------
__global__ void k_multihop(const float* __restrict__ tp, const int* __restrict__ head,
                           const int* __restrict__ tail, const int* __restrict__ dist,
                           const float* __restrict__ cntm, float* __restrict__ cprob) {
  __shared__ float ns[M_], sm[M_], tot[M_], red[256];
  const int row = blockIdx.x;        // b*L + l
  const int b = row >> 7;
  const int tid = threadIdx.x;
  for (int m = tid; m < M_; m += 256) {
    float init = (dist[b * M_ + m] == 0) ? 1.0f : 0.0f;
    ns[m] = init;
    tot[m] = init * -100000.0f;
  }
  __syncthreads();
  for (int hop = 0; hop < 2; ++hop) {
    for (int m = tid; m < M_; m += 256) sm[m] = 0.0f;
    __syncthreads();
    for (int t = tid; t < MT_; t += 256) {
      int h = head[b * MT_ + t], ta = tail[b * MT_ + t];
      float u = ns[h] * 0.8f + tp[(size_t)row * MT_ + t];
      atomicAdd(&sm[ta], u);
    }
    __syncthreads();
    for (int m = tid; m < M_; m += 256) {
      float c = cntm[b * M_ + m];
      c = c < 1.0f ? 1.0f : c;
      float v = sm[m] / c;
      ns[m] = v;
      int d = dist[b * M_ + m];
      float decay = (d == 0) ? 1.0f : (d == 1 ? 0.8f : 0.64f);   // 0.8^d, d in {0,1,2}
      tot[m] += v * decay;
    }
    __syncthreads();
  }
  // softmax over tot[0..M_-1]
  float mx = -1e30f;
  for (int m = tid; m < M_; m += 256) mx = fmaxf(mx, tot[m]);
  red[tid] = mx; __syncthreads();
  for (int s = 128; s > 0; s >>= 1) { if (tid < s) red[tid] = fmaxf(red[tid], red[tid + s]); __syncthreads(); }
  mx = red[0]; __syncthreads();
  float lsum = 0.0f;
  for (int m = tid; m < M_; m += 256) lsum += expf(tot[m] - mx);
  red[tid] = lsum; __syncthreads();
  for (int s = 128; s > 0; s >>= 1) { if (tid < s) red[tid] += red[tid + s]; __syncthreads(); }
  float inv = 1.0f / red[0];
  __syncthreads();
  for (int m = tid; m < M_; m += 256)
    cprob[(size_t)row * M_ + m] = expf(tot[m] - mx) * inv;
}

// probs = gate*cpv + (1-gate)*softmax(logits), in place on d_out
__global__ void k_final(float* __restrict__ out, const float* __restrict__ stats,
                        const float* __restrict__ gate, const float* __restrict__ cprob,
                        const int* __restrict__ vmap, const int* __restrict__ mmask) {
  int v = blockIdx.x * 256 + threadIdx.x;
  if (v >= V_) return;
  int row = blockIdx.y;
  float mx  = stats[row * 2];
  float inv = 1.0f / stats[row * 2 + 1];
  size_t idx = (size_t)row * V_ + v;
  float lm = expf(out[idx] - mx) * inv;
  float g  = gate[row];
  float cp = (mmask[v] != 0) ? cprob[(size_t)row * M_ + vmap[v]] : 0.0f;
  out[idx] = g * cp + (1.0f - g) * lm;
}

// ---------------------------------------------------------------------------
extern "C" void kernel_launch(void* const* d_in, const int* in_sizes, int n_in,
                              void* d_out, int out_size, void* d_ws, size_t ws_size,
                              hipStream_t stream) {
  const float* hidden = (const float*)d_in[0];
  const float* wte    = (const float*)d_in[1];
  const float* rel    = (const float*)d_in[2];
  // Only the last hop (l = NUM_HOPS-1 = 1) survives in the reference loop.
  const float* Ws1 = (const float*)d_in[3] + (size_t)E_ * E_;
  const float* Wn1 = (const float*)d_in[4] + (size_t)E_ * E_;
  const float* Wr1 = (const float*)d_in[5] + (size_t)E_ * E_;
  const float* Wt  = (const float*)d_in[6];
  const float* gw  = (const float*)d_in[7];
  const float* gb  = (const float*)d_in[8];
  const int* cid   = (const int*)d_in[9];
  const int* rid   = (const int*)d_in[10];
  const int* head  = (const int*)d_in[11];
  const int* tail  = (const int*)d_in[12];
  const int* lab   = (const int*)d_in[13];
  const int* dist  = (const int*)d_in[14];
  const int* vmap  = (const int*)d_in[15];
  const int* mmask = (const int*)d_in[16];
  float* out = (float*)d_out;

  // -------- workspace layout: f32 region, then f16 region ----------------
  float* ws    = (float*)d_ws;
  float* cr    = ws;                                 // B*M*E
  float* rr    = cr   + (size_t)B_ * M_  * E_;       // B*Mt*E
  float* upd   = rr   + (size_t)B_ * MT_ * E_;       // B*M*E   (zeroed)
  float* cntg  = upd  + (size_t)B_ * M_  * E_;       // B*M     (zeroed)
  float* cntm  = cntg + (size_t)B_ * M_;             // B*M     (zeroed)
  float* tp    = cntm + (size_t)B_ * M_;             // B*L*Mt
  float* cprob = tp   + (size_t)B_ * L_  * MT_;      // B*L*M
  float* stats = cprob+ (size_t)B_ * L_  * M_;       // B*L*2
  float* gate  = stats+ (size_t)BL_ * 2;             // B*L

  _Float16* h16    = (_Float16*)(gate + BL_);            // B*L*E
  _Float16* wte_h  = h16    + (size_t)BL_ * E_;          // V*E   (77MB, L2-resident)
  _Float16* node_h = wte_h  + (size_t)V_  * E_;          // B*M*E
  _Float16* relh_h = node_h + (size_t)B_ * M_  * E_;     // B*Mt*E
  _Float16* tl_h   = relh_h + (size_t)B_ * MT_ * E_;     // B*Mt*E
  _Float16* Ws1_h  = tl_h   + (size_t)B_ * MT_ * E_;     // E*E
  _Float16* Wn1_h  = Ws1_h  + (size_t)E_ * E_;           // E*E
  _Float16* Wr1_h  = Wn1_h  + (size_t)E_ * E_;           // E*E
  _Float16* Wt_h   = Wr1_h  + (size_t)E_ * E_;           // E*3E

  const int zero_n = B_ * M_ * E_ + 2 * B_ * M_;     // upd + cntg + cntm
  k_zero<<<(zero_n + 255) / 256, 256, 0, stream>>>(upd, zero_n);

  // f32 -> f16 staging (wte dominates: 50257*768 = 38.6M elems)
  k_cvt<<<(BL_ * E_ + 255) / 256, 256, 0, stream>>>(hidden, h16, BL_ * E_);
  k_cvt<<<((int)((size_t)V_ * E_) + 255) / 256, 256, 0, stream>>>(wte, wte_h, V_ * E_);
  k_cvt<<<(E_ * E_ + 255) / 256, 256, 0, stream>>>(Ws1, Ws1_h, E_ * E_);
  k_cvt<<<(E_ * E_ + 255) / 256, 256, 0, stream>>>(Wn1, Wn1_h, E_ * E_);
  k_cvt<<<(E_ * E_ + 255) / 256, 256, 0, stream>>>(Wr1, Wr1_h, E_ * E_);
  k_cvt<<<(3 * E_ * E_ + 255) / 256, 256, 0, stream>>>(Wt, Wt_h, 3 * E_ * E_);

  k_gather<<<(B_ * M_  * 192 + 255) / 256, 256, 0, stream>>>(wte, cid, cr, B_ * M_);
  k_gather<<<(B_ * MT_ * 192 + 255) / 256, 256, 0, stream>>>(rel, rid, rr, B_ * MT_);
  k_cnt<<<(B_ * MT_ + 255) / 256, 256, 0, stream>>>(head, tail, cntg, cntm);
  k_scatter<<<(B_ * MT_ * E_ + 255) / 256, 256, 0, stream>>>(cr, rr, head, tail, upd);

  k_node<<<dim3(M_ / 16, E_ / 16, B_), 32, 0, stream>>>(cr, upd, Ws1_h, Wn1_h, cntg, node_h);
  k_relh<<<dim3(MT_ / 16, E_ / 16, B_), 32, 0, stream>>>(rr, Wr1_h, relh_h);
  k_tl  <<<dim3(MT_ / 16, E_ / 16, B_), 32, 0, stream>>>(node_h, relh_h, head, tail, Wt_h, tl_h);
  k_tp  <<<dim3(MT_ / 16, L_ / 16, B_), 32, 0, stream>>>(h16, tl_h, lab, tp);

  k_logits<<<dim3((V_ + 63) / 64, BL_ / 16), 128, 0, stream>>>(h16, wte_h, out);
  k_stats<<<BL_, 256, 0, stream>>>(out, stats);
  k_gate<<<(BL_ + 255) / 256, 256, 0, stream>>>(hidden, gw, gb, gate);
  k_multihop<<<BL_, 256, 0, stream>>>(tp, head, tail, dist, cntm, cprob);
  k_final<<<dim3((V_ + 255) / 256, BL_), 256, 0, stream>>>(out, stats, gate, cprob, vmap, mmask);
}
